// SimpleGATLayer_53128745451825
// MI455X (gfx1250) — matrix operand order repaired and checked by reading, not compiled, not run on hardware
//
#include <hip/hip_runtime.h>

// GAT layer: B=8, S=1024, F_IN=768, H=8, D=96 (H*D=768)
#define B_   8
#define S_   1024
#define FIN_ 768
#define H_   8
#define D_   96
#define HD_  768
#define ALPHA_ 0.2f
#define NEG_   (-1e9f)

#define AS1 __attribute__((address_space(1)))
#define AS3 __attribute__((address_space(3)))

typedef __attribute__((ext_vector_type(16))) __bf16 v16bf;
typedef __attribute__((ext_vector_type(8)))  __bf16 v8bf;
typedef __attribute__((ext_vector_type(8)))  float  v8f;
typedef __attribute__((ext_vector_type(4)))  int    v4i;

union AB16 { v16bf v; v8bf h[2]; unsigned short u[16]; };

#if __has_builtin(__builtin_amdgcn_global_load_async_to_lds_b128)
#define HAVE_ASYNC_LDS 1
#else
#define HAVE_ASYNC_LDS 0
#endif

__device__ __forceinline__ void wait_async0() {
#if __has_builtin(__builtin_amdgcn_s_wait_asynccnt)
  __builtin_amdgcn_s_wait_asynccnt(0);
#else
  asm volatile("s_wait_asynccnt 0x0" ::: "memory");
#endif
}

// f32 -> bf16, round-to-nearest-even (bit twiddle; avoids __bf16 conversion deps)
__device__ __forceinline__ unsigned short f2bf(float f) {
  unsigned int u = __float_as_uint(f);
  unsigned int r = (u + 0x7FFFu + ((u >> 16) & 1u)) >> 16;
  return (unsigned short)r;
}

// ---------------------------------------------------------------------------
// Kernel 1: Wh = h @ W   (M=8192, N=768, K=768), bf16 WMMA, f32 accumulate.
// Block tile 64(M) x 128(N), K-step 32. 256 threads = 8 waves (2x4 grid),
// each wave owns 32x32 = 2x2 WMMA accumulators.
// ---------------------------------------------------------------------------
__global__ __launch_bounds__(256) void gat_wh_gemm(const float* __restrict__ h,
                                                   const float* __restrict__ W,
                                                   float* __restrict__ Wh) {
  __shared__ __align__(16) unsigned short Alds[64 * 40];   // A: 64 x 32 bf16
  __shared__ __align__(16) unsigned short Btld[128 * 40];  // B transposed: 128 x 32 bf16

  const int tid   = threadIdx.x;
  const int lane  = tid & 31;
  const int wave  = tid >> 5;
  const int m0    = blockIdx.x * 64;
  const int n0    = blockIdx.y * 128;
  const int wm    = (wave & 1) * 32;
  const int wn    = (wave >> 1) * 32;
  const int ln    = lane & 15;
  const int khalf = lane >> 4;

  v8f acc[2][2];
#pragma unroll
  for (int i = 0; i < 2; ++i)
#pragma unroll
    for (int j = 0; j < 2; ++j)
#pragma unroll
      for (int e = 0; e < 8; ++e) acc[i][j][e] = 0.0f;

  for (int k0 = 0; k0 < FIN_; k0 += 32) {
    __syncthreads();
#pragma unroll
    for (int i = 0; i < 8; ++i) {        // stage A 64x32 (coalesced along K)
      int idx = tid + i * 256;
      int r = idx >> 5, c = idx & 31;
      Alds[r * 40 + c] = f2bf(h[(m0 + r) * FIN_ + k0 + c]);
    }
#pragma unroll
    for (int i = 0; i < 16; ++i) {       // stage B 32x128 transposed (N-major)
      int idx = tid + i * 256;
      int r = idx >> 7, c = idx & 127;
      Btld[c * 40 + r] = f2bf(W[(k0 + r) * HD_ + n0 + c]);
    }
    if (k0 + 32 < FIN_) {                // global_prefetch_b8 hints for next tiles
      __builtin_prefetch(h + (m0 + (tid >> 2)) * FIN_ + k0 + 32);
      __builtin_prefetch(W + (k0 + 32 + (tid >> 4)) * HD_ + n0 + (tid & 15) * 8);
    }
    __syncthreads();

    // A frag (16x32): lanes 0-15 hold K 0..7 & 16..23 ; lanes 16-31 K 8..15 & 24..31
    AB16 aF[2], bF[2];
#pragma unroll
    for (int m16 = 0; m16 < 2; ++m16) {
      int row = wm + m16 * 16 + ln;
      aF[m16].h[0] = *reinterpret_cast<const v8bf*>(&Alds[row * 40 + khalf * 8]);
      aF[m16].h[1] = *reinterpret_cast<const v8bf*>(&Alds[row * 40 + 16 + khalf * 8]);
    }
    // B frag (32x16): N = lane&15; lanes 0-15 hold K 0..7 & 16..23 ; lanes 16-31 K 8..15 & 24..31
#pragma unroll
    for (int n16 = 0; n16 < 2; ++n16) {
      int col = wn + n16 * 16 + ln;
      bF[n16].h[0] = *reinterpret_cast<const v8bf*>(&Btld[col * 40 + khalf * 8]);
      bF[n16].h[1] = *reinterpret_cast<const v8bf*>(&Btld[col * 40 + 16 + khalf * 8]);
    }
#pragma unroll
    for (int m16 = 0; m16 < 2; ++m16)
#pragma unroll
      for (int n16 = 0; n16 < 2; ++n16)
        acc[m16][n16] = __builtin_amdgcn_wmma_f32_16x16x32_bf16(
            false, aF[m16].v, false, bF[n16].v, (short)0, acc[m16][n16], false, false);
  }

#pragma unroll
  for (int m16 = 0; m16 < 2; ++m16)
#pragma unroll
    for (int n16 = 0; n16 < 2; ++n16)
#pragma unroll
      for (int e = 0; e < 8; ++e) {
        int row = m0 + wm + m16 * 16 + e + 8 * khalf;
        int col = n0 + wn + n16 * 16 + ln;
        Wh[row * HD_ + col] = acc[m16][n16][e];
      }
}

// ---------------------------------------------------------------------------
// Kernel 2: src/dst per (b,s,h): 96-wide dot with a[h,:D] / a[h,D:]
// ---------------------------------------------------------------------------
__global__ __launch_bounds__(256) void gat_srcdst(const float* __restrict__ Wh,
                                                  const float* __restrict__ a,
                                                  float* __restrict__ srcO,
                                                  float* __restrict__ dstO) {
  int idx = blockIdx.x * 256 + threadIdx.x;  // B*S*H = 65536
  if (idx >= B_ * S_ * H_) return;
  int hh = idx & 7;
  int bs = idx >> 3;
  const float* wrow = Wh + (size_t)bs * HD_ + hh * D_;
  const float* av   = a + hh * 2 * D_;
  float s = 0.f, d = 0.f;
#pragma unroll 4
  for (int t = 0; t < D_; ++t) {
    float w = wrow[t];
    s += w * av[t];
    d += w * av[D_ + t];
  }
  srcO[idx] = s;
  dstO[idx] = d;
}

// ---------------------------------------------------------------------------
// Kernel 2b: WhB[bh][d][s] = bf16(Wh[b,s,h,d])  (d-major per head, so the
// attention kernel's V tiles are contiguous and can be DMA'd into LDS without
// any VALU conversion). Tiled through LDS so both sides are coalesced.
// ---------------------------------------------------------------------------
__global__ __launch_bounds__(256) void gat_transpose(const float* __restrict__ Wh,
                                                     unsigned short* __restrict__ WhB) {
  __shared__ unsigned short T[96 * 72];  // [d][s] tile 96 x 64 (stride 72 vs conflicts)
  const int tid = threadIdx.x;
  const int bh  = blockIdx.x;            // B*H = 64
  const int b   = bh >> 3, hh = bh & 7;
  const int s0  = blockIdx.y * 64;
  for (int idx = tid; idx < 64 * 96; idx += 256) {   // read coalesced along d
    int s = idx / 96, d = idx - s * 96;
    T[d * 72 + s] = f2bf(Wh[((size_t)b * S_ + s0 + s) * HD_ + hh * D_ + d]);
  }
  __syncthreads();
  for (int idx = tid; idx < 96 * 64; idx += 256) {   // write coalesced along s
    int d = idx >> 6, s = idx & 63;
    WhB[((size_t)(bh * 96 + d) << 10) + s0 + s] = T[d * 72 + s];
  }
}

// ---------------------------------------------------------------------------
// Kernel 3: flash-attention style GAT attention.
// Block = 128 threads (4 waves) owns (b,h, 64 queries); each wave a 16-query
// tile. Loop over 32-key tiles. V tiles (bf16, d-major) are DMA'd into LDS via
// GLOBAL_LOAD_ASYNC_TO_LDS_B128 with double buffering (copy of tile j+1
// overlaps compute of tile j); falls back to synchronous staging if the async
// builtins are unavailable. Scores are built directly in the WMMA A-fragment
// layout; online softmax; 6 WMMAs (16x96) per key tile.
// Query-side mask bias is a per-row constant => softmax-invariant => dropped.
// ---------------------------------------------------------------------------
__global__ __launch_bounds__(128) void gat_attn(const unsigned short* __restrict__ WhB,
                                                const float* __restrict__ srcB,
                                                const float* __restrict__ dstB,
                                                const int* __restrict__ mask,
                                                float* __restrict__ out) {
  // Vt[buf][d][k]: 96 rows x 32 keys bf16, row stride 40 halves (80B: 16B-aligned
  // chunks, 20-dword row stride => conflict-free 16-lane fragment reads)
  __shared__ __align__(16) unsigned short VtBuf[2][96 * 40];
  __shared__ float dstT[2][32];
  __shared__ float biasT[2][32];

  const int tid   = threadIdx.x;
  const int lane  = tid & 31;
  const int wave  = tid >> 5;
  const int bh    = blockIdx.x;           // B*H = 64
  const int b     = bh >> 3, hh = bh & 7;
  const int q0    = blockIdx.y * 64 + wave * 16;
  const int ln    = lane & 15;            // query row M in A-frag layout
  const int khalf = lane >> 4;

  const float srcRow = srcB[((size_t)b * S_ + q0 + ln) * H_ + hh];
  float m = -INFINITY, l = 0.f;
  v8f acc[6];
#pragma unroll
  for (int nt = 0; nt < 6; ++nt)
#pragma unroll
    for (int e = 0; e < 8; ++e) acc[nt][e] = 0.0f;

  // stage dst + key-mask-bias for tile jt into slot `buf`
  auto stage_dst = [&](int jt, int buf) {
    if (tid < 32) {
      int j = jt * 32 + tid;
      dstT[buf][tid]  = dstB[((size_t)b * S_ + j) * H_ + hh];
      biasT[buf][tid] = mask[b * S_ + j] ? 0.f : NEG_;
    }
  };

#if HAVE_ASYNC_LDS
  // DMA a 32-key V tile (96 x 32 bf16 = 6KB) into LDS: 384 16B chunks, 3/thread.
  auto issue_tile = [&](int jt, int buf) {
    int j0 = jt * 32;
#pragma unroll
    for (int i = 0; i < 3; ++i) {
      int id = tid + i * 128;
      int d = id >> 2, c = id & 3;
      const unsigned short* g = WhB + ((size_t)(bh * 96 + d) << 10) + j0 + c * 8;
      unsigned short* lp = &VtBuf[buf][d * 40 + c * 8];
      __builtin_amdgcn_global_load_async_to_lds_b128((AS1 v4i*)g, (AS3 v4i*)lp, 0, 0);
    }
  };
  issue_tile(0, 0);
  stage_dst(0, 0);
#endif

  for (int jt = 0; jt < S_ / 32; ++jt) {
#if HAVE_ASYNC_LDS
    const int cur = jt & 1;
    wait_async0();        // my copies for tile jt landed
    __syncthreads();      // all waves' copies landed; all done reading buf[cur^1]
    if (jt + 1 < S_ / 32) {
      issue_tile(jt + 1, cur ^ 1);   // DMA next tile, overlapped with compute below
      stage_dst(jt + 1, cur ^ 1);    // visible after next iteration's barrier
    }
#else
    const int cur = 0;
    __syncthreads();      // protect previous tile's LDS reads
    for (int idx = tid; idx < 96 * 32; idx += 128) {  // sync staging fallback
      int d = idx >> 5, k = idx & 31;
      VtBuf[0][d * 40 + k] = WhB[((size_t)(bh * 96 + d) << 10) + jt * 32 + k];
    }
    stage_dst(jt, 0);
    __syncthreads();
#endif

    // Scores in A-fragment layout: lane slot t -> key k
    float p[16];
    float tmax = -INFINITY;
#pragma unroll
    for (int t = 0; t < 16; ++t) {
      int k = khalf * 8 + t + (t >= 8 ? 8 : 0);
      float x = srcRow + dstT[cur][k];
      x = x > 0.f ? x : ALPHA_ * x;     // leaky relu
      x += biasT[cur][k];               // key mask bias (after lrelu, as reference)
      p[t] = x;
      tmax = fmaxf(tmax, x);
    }
    tmax = fmaxf(tmax, __shfl_xor(tmax, 16, 32));  // combine same-row lane halves
    float mn    = fmaxf(m, tmax);
    float scale = __expf(m - mn);                  // first tile: exp(-inf)=0
    float lsum  = 0.f;
    AB16 aF;
#pragma unroll
    for (int t = 0; t < 16; ++t) {
      float ev = __expf(p[t] - mn);
      lsum += ev;
      aF.u[t] = f2bf(ev);
    }
    lsum += __shfl_xor(lsum, 16, 32);
    l = l * scale + lsum;
    m = mn;

    float sc[8];  // per-row rescale broadcast into C-layout rows (row = e + 8*khalf)
#pragma unroll
    for (int e = 0; e < 8; ++e) sc[e] = __shfl(scale, e + 8 * khalf, 32);

#pragma unroll
    for (int nt = 0; nt < 6; ++nt) {
#pragma unroll
      for (int e = 0; e < 8; ++e) acc[nt][e] *= sc[e];
      AB16 bF;  // B frag: col N = lane&15 of this d-tile, K half by khalf
      int off = (nt * 16 + ln) * 40 + khalf * 8;
      bF.h[0] = *reinterpret_cast<const v8bf*>(&VtBuf[cur][off]);
      bF.h[1] = *reinterpret_cast<const v8bf*>(&VtBuf[cur][off + 16]);
      acc[nt] = __builtin_amdgcn_wmma_f32_16x16x32_bf16(
          false, aF.v, false, bF.v, (short)0, acc[nt], false, false);
    }
  }

  float rl[8];
#pragma unroll
  for (int e = 0; e < 8; ++e) rl[e] = 1.f / __shfl(l, e + 8 * khalf, 32);
#pragma unroll
  for (int nt = 0; nt < 6; ++nt)
#pragma unroll
    for (int e = 0; e < 8; ++e) {
      int row = q0 + e + 8 * khalf;
      out[((size_t)b * S_ + row) * HD_ + hh * D_ + nt * 16 + ln] = acc[nt][e] * rl[e];
    }
}

// ---------------------------------------------------------------------------
extern "C" void kernel_launch(void* const* d_in, const int* in_sizes, int n_in,
                              void* d_out, int out_size, void* d_ws, size_t ws_size,
                              hipStream_t stream) {
  const float* h    = (const float*)d_in[0];   // [B,S,F_IN]
  const float* W    = (const float*)d_in[1];   // [F_IN, H*D]
  const float* a    = (const float*)d_in[2];   // [H, 2*D]
  const int*   mask = (const int*)d_in[3];     // [B,S]
  float* out = (float*)d_out;                  // [B,S,H*D]

  // workspace: Wh f32 (24MB) + src/dst (0.5MB) + WhB bf16 (12.6MB) ~= 37.2MB
  float* Wh   = (float*)d_ws;
  float* srcB = Wh + (size_t)B_ * S_ * HD_;
  float* dstB = srcB + (size_t)B_ * S_ * H_;
  unsigned short* WhB = (unsigned short*)(dstB + (size_t)B_ * S_ * H_);

  dim3 g1((B_ * S_) / 64, HD_ / 128);          // 128 x 6 blocks
  gat_wh_gemm<<<g1, 256, 0, stream>>>(h, W, Wh);

  gat_srcdst<<<dim3((B_ * S_ * H_) / 256), 256, 0, stream>>>(Wh, a, srcB, dstB);

  gat_transpose<<<dim3(B_ * H_, S_ / 64), 256, 0, stream>>>(Wh, WhB);

  gat_attn<<<dim3(B_ * H_, S_ / 64), 128, 0, stream>>>(WhB, srcB, dstB, mask, out);
}
